// FlamingoMaskedCrossAttention_9174050144688
// MI455X (gfx1250) — matrix operand order repaired and checked by reading, not compiled, not run on hardware
//
#include <hip/hip_runtime.h>

typedef __bf16 bf16;
typedef __attribute__((ext_vector_type(16))) __bf16 v16bf;
typedef __attribute__((ext_vector_type(8)))  float  v8f;
typedef __attribute__((ext_vector_type(4)))  unsigned int u32x4;
typedef int v4i __attribute__((vector_size(16)));   // matches builtin prototype

#define BB 4
#define TT 2048
#define DD 2048
#define DVV 1024
#define HH 16
#define DHH 64
#define INNERD 1024
#define JJ 512   // T_IMG * N_LAT

#if __has_builtin(__builtin_amdgcn_global_load_async_to_lds_b128)
#define USE_ASYNC_LDS 1
#else
#define USE_ASYNC_LDS 0
#endif

union FragU { u32x4 u4[2]; v16bf v; };

__device__ __forceinline__ bf16 f2bf(float f) {
    unsigned int u = __float_as_uint(f);
    unsigned int r = (u + 0x7FFFu + ((u >> 16) & 1u)) >> 16;
    unsigned short h = (unsigned short)r;
    return __builtin_bit_cast(bf16, h);
}

__device__ __forceinline__ void async_b128(const void* g, void* l) {
#if USE_ASYNC_LDS
    __builtin_amdgcn_global_load_async_to_lds_b128(
        (__attribute__((address_space(1))) v4i*)g,
        (__attribute__((address_space(3))) v4i*)l, 0, 0);
#else
    (void)g; (void)l;
#endif
}

__device__ __forceinline__ void wait_async0() {
#if __has_builtin(__builtin_amdgcn_s_wait_asynccnt)
    __builtin_amdgcn_s_wait_asynccnt(0);
#else
    asm volatile("s_wait_asynccnt 0x0" ::: "memory");
#endif
}

// ---------------- LayerNorm -> bf16 ----------------
__global__ __launch_bounds__(256)
void ln_rows_kernel(const float* __restrict__ x, const float* __restrict__ gamma,
                    const float* __restrict__ beta, bf16* __restrict__ xn) {
    int row = blockIdx.x;                 // B*T rows
    int tid = threadIdx.x;
    const float* xr = x + (size_t)row * DD;
    float vals[8];
    float s = 0.f, ss = 0.f;
#pragma unroll
    for (int i = 0; i < 8; i++) {
        float v = xr[tid + i * 256];
        vals[i] = v; s += v; ss += v * v;
    }
    __shared__ float rs[256], rss[256];
    rs[tid] = s; rss[tid] = ss;
    __syncthreads();
    for (int st = 128; st > 0; st >>= 1) {
        if (tid < st) { rs[tid] += rs[tid + st]; rss[tid] += rss[tid + st]; }
        __syncthreads();
    }
    float mu  = rs[0] * (1.0f / DD);
    float var = rss[0] * (1.0f / DD) - mu * mu;
    float inv = rsqrtf(var + 1e-5f);
    bf16* xo = xn + (size_t)row * DD;
#pragma unroll
    for (int i = 0; i < 8; i++) {
        int c = tid + i * 256;
        xo[c] = f2bf((vals[i] - mu) * inv * gamma[c] + beta[c]);
    }
}

// ---------------- transpose fp32 (RxC) -> bf16 (CxR) ----------------
__global__ __launch_bounds__(256)
void transpose_f32_to_bf16_kernel(const float* __restrict__ in, bf16* __restrict__ out,
                                  int R, int C) {
    __shared__ float tile[32][33];
    int bc = blockIdx.x * 32;   // column tile
    int br = blockIdx.y * 32;   // row tile
    int tx = threadIdx.x & 31;
    int ty = threadIdx.x >> 5;  // 0..7
#pragma unroll
    for (int i = 0; i < 32; i += 8)
        tile[ty + i][tx] = in[(size_t)(br + ty + i) * C + bc + tx];
    __syncthreads();
#pragma unroll
    for (int i = 0; i < 32; i += 8)
        out[(size_t)(bc + ty + i) * R + br + tx] = f2bf(tile[tx][ty + i]);
}

// ---------------- elementwise fp32 -> bf16 ----------------
__global__ __launch_bounds__(256)
void cvt_bf16_kernel(const float* __restrict__ in, bf16* __restrict__ out, int n) {
    int i = (blockIdx.x * 256 + threadIdx.x) * 4;
    if (i + 3 < n) {
#pragma unroll
        for (int k = 0; k < 4; k++) out[i + k] = f2bf(in[i + k]);
    }
}

// ---------------- media-location scan -> per-batch block starts ----------------
__global__ void starts_kernel(const unsigned char* __restrict__ loc, int* __restrict__ starts) {
    int b = blockIdx.x;
    if (threadIdx.x == 0) {
        int tt = 0;
        starts[b * 10 + 0] = 0;
        for (int m = 1; m <= 9; m++) starts[b * 10 + m] = TT;
        for (int t = 0; t < TT; t++) {
            if (loc[b * TT + t]) {
                tt++;
                if (tt <= 8) starts[b * 10 + tt] = t;
            }
        }
        starts[b * 10 + 9] = TT;
    }
}

// ---------------- bf16 WMMA GEMM: C(MxN) = A(MxK) * Bt(NxK)^T ----------------
// M%128==0, N%128==0, K%32==0. Double-buffered LDS, async global->LDS staging
// (ASYNCcnt) when the toolchain exposes it, else reg-staged double buffering.
__device__ __forceinline__ v16bf load_frag_a(const bf16* srow, int lane) {
    const unsigned int* p = (const unsigned int*)srow;
    int o = (lane >> 4) << 2;            // 0 or 4 dwords
    FragU f;
    f.u4[0] = *(const u32x4*)(p + o);
    f.u4[1] = *(const u32x4*)(p + o + 8);
    return f.v;
}
__device__ __forceinline__ v16bf load_frag_b(const bf16* srow, int lane) {
    const unsigned int* p = (const unsigned int*)srow;
    int o = (lane >> 4) << 2;            // 0 or 4 dwords
    FragU f;
    f.u4[0] = *(const u32x4*)(p + o);
    f.u4[1] = *(const u32x4*)(p + o + 8);
    return f.v;
}

__global__ __launch_bounds__(256)
void gemm_bf16_wmma_kernel(const bf16* __restrict__ A, const bf16* __restrict__ Bt,
                           float* __restrict__ C, int M, int N, int K) {
    constexpr int LDS_K = 40;            // 32 + 8 bf16 pad (keeps 16B alignment)
    constexpr int BUF   = 128 * LDS_K;   // one matrix buffer (bf16 elems)
    __shared__ bf16 As[2 * BUF];
    __shared__ bf16 Bs[2 * BUF];

    int tid  = threadIdx.x;
    int lane = tid & 31;
    int wave = tid >> 5;                 // 0..7
    int wm   = (wave & 1) * 64;          // wave M offset in tile
    int wn   = (wave >> 1) * 32;         // wave N offset in tile
    int bm   = blockIdx.y * 128;
    int bn   = blockIdx.x * 128;

    int srow = tid >> 1;                 // 0..127 staging row
    int sseg = (tid & 1) << 4;           // 0 or 16 bf16

    const bf16* agBase = A  + (size_t)(bm + srow) * K + sseg;
    const bf16* bgBase = Bt + (size_t)(bn + srow) * K + sseg;
    bf16* asBase = As + srow * LDS_K + sseg;
    bf16* bsBase = Bs + srow * LDS_K + sseg;

    v8f acc[4][2];
#pragma unroll
    for (int i = 0; i < 4; i++)
#pragma unroll
        for (int j = 0; j < 2; j++)
            acc[i][j] = v8f{0.f, 0.f, 0.f, 0.f, 0.f, 0.f, 0.f, 0.f};

    int nk = K >> 5;

    // ---- prologue: stage slice 0 into buffer 0 ----
#if USE_ASYNC_LDS
    async_b128(agBase,      asBase);
    async_b128(agBase + 8,  asBase + 8);
    async_b128(bgBase,      bsBase);
    async_b128(bgBase + 8,  bsBase + 8);
    wait_async0();
#else
    {
        u32x4 a0 = *(const u32x4*)agBase;
        u32x4 a1 = *(const u32x4*)(agBase + 8);
        u32x4 b0 = *(const u32x4*)bgBase;
        u32x4 b1 = *(const u32x4*)(bgBase + 8);
        *(u32x4*)asBase       = a0;
        *(u32x4*)(asBase + 8) = a1;
        *(u32x4*)bsBase       = b0;
        *(u32x4*)(bsBase + 8) = b1;
    }
#endif
    __syncthreads();

    for (int kt = 0; kt < nk; kt++) {
        int cur = (kt & 1) * BUF;
        int nxt = ((kt + 1) & 1) * BUF;
        const bf16* Acur = As + cur;
        const bf16* Bcur = Bs + cur;
        int knext = (kt + 1) << 5;
        bool more = (kt + 1) < nk;

#if USE_ASYNC_LDS
        if (more) {   // DMA next slice into the other buffer while we compute
            async_b128(agBase + knext,      asBase + nxt);
            async_b128(agBase + knext + 8,  asBase + nxt + 8);
            async_b128(bgBase + knext,      bsBase + nxt);
            async_b128(bgBase + knext + 8,  bsBase + nxt + 8);
        }
#else
        u32x4 a0, a1, b0, b1;
        if (more) {   // issue global loads early; consumed after the WMMAs
            a0 = *(const u32x4*)(agBase + knext);
            a1 = *(const u32x4*)(agBase + knext + 8);
            b0 = *(const u32x4*)(bgBase + knext);
            b1 = *(const u32x4*)(bgBase + knext + 8);
        }
#endif

        v16bf af[4], bfr[2];
#pragma unroll
        for (int ti = 0; ti < 4; ti++)
            af[ti] = load_frag_a(Acur + (wm + ti * 16 + (lane & 15)) * LDS_K, lane);
#pragma unroll
        for (int tj = 0; tj < 2; tj++)
            bfr[tj] = load_frag_b(Bcur + (wn + tj * 16 + (lane & 15)) * LDS_K, lane);

#pragma unroll
        for (int ti = 0; ti < 4; ti++)
#pragma unroll
            for (int tj = 0; tj < 2; tj++)
                acc[ti][tj] = __builtin_amdgcn_wmma_f32_16x16x32_bf16(
                    false, af[ti], false, bfr[tj], (short)0, acc[ti][tj], false, false);

#if USE_ASYNC_LDS
        if (more) wait_async0();
#else
        if (more) {
            *(u32x4*)(asBase + nxt)     = a0;
            *(u32x4*)(asBase + nxt + 8) = a1;
            *(u32x4*)(bsBase + nxt)     = b0;
            *(u32x4*)(bsBase + nxt + 8) = b1;
        }
#endif
        __syncthreads();
    }

    // ---- epilogue: pointer-increment stores (no per-element u64 multiplies) ----
    int ln15 = lane & 15, lhi = lane >> 4;
#pragma unroll
    for (int ti = 0; ti < 4; ti++) {
#pragma unroll
        for (int tj = 0; tj < 2; tj++) {
            float* cp = C + (size_t)(bm + wm + ti * 16 + lhi * 8) * N
                          + (bn + wn + tj * 16 + ln15);
#pragma unroll
            for (int r = 0; r < 8; r++) {
                *cp = acc[ti][tj][r];
                cp += N;
            }
        }
    }
}

// ---------------- masked cross-attention (64 keys per query) ----------------
__global__ __launch_bounds__(256)
void flamingo_attn_kernel(const float* __restrict__ q, const float* __restrict__ kv,
                          const int* __restrict__ starts, bf16* __restrict__ attout) {
    int h = blockIdx.x;      // head
    int m = blockIdx.y;      // media block index 0..8 (0 => no media: zero rows)
    int b = blockIdx.z;
    int t0 = starts[b * 10 + m];
    int t1 = starts[b * 10 + m + 1];
    int tid = threadIdx.x;

    if (m == 0) {
        int tq = tid >> 6, d = tid & 63;
        for (int t = t0; t < t1; t += 4) {
            int tt = t + tq;
            if (tt < t1)
                attout[(size_t)(b * TT + tt) * INNERD + h * DHH + d] = f2bf(0.f);
        }
        return;
    }

    __shared__ float ksh[64][65];
    __shared__ float vsh[64][65];
    __shared__ float qsh[4][64];
    __shared__ float ssh[4][64];

    {   // stage K/V for this (b, media block m, head h): 64 keys x 64 dims
        int j = tid >> 2;
        int ds = (tid & 3) * 16;
        const float* krow = kv + (size_t)(b * JJ + (m - 1) * 64 + j) * (2 * INNERD) + h * DHH;
#pragma unroll
        for (int d = 0; d < 16; d++) {
            ksh[j][ds + d] = krow[ds + d];
            vsh[j][ds + d] = krow[INNERD + ds + d];
        }
    }
    __syncthreads();

    int tq = tid >> 6;       // 0..3 query slot
    int lj = tid & 63;       // key index / output dim
    for (int t = t0; t < t1; t += 4) {
        {   // load 4 query rows (pre-scaled)
            int qr = tid >> 6, qd = tid & 63;
            if (t + qr < t1)
                qsh[qr][qd] = q[(size_t)(b * TT + t + qr) * INNERD + h * DHH + qd] * 0.125f;
        }
        __syncthreads();

        float s = 0.f;
#pragma unroll
        for (int d = 0; d < 64; d++) s += qsh[tq][d] * ksh[lj][d];
        ssh[tq][lj] = s;
        __syncthreads();

        float mx = -3.402823466e38f;
#pragma unroll
        for (int jj = 0; jj < 64; jj++) mx = fmaxf(mx, ssh[tq][jj]);
        __syncthreads();                 // all reads done before overwrite

        ssh[tq][lj] = __expf(s - mx);
        __syncthreads();

        float Z = 0.f, o = 0.f;
#pragma unroll
        for (int jj = 0; jj < 64; jj++) {
            float w = ssh[tq][jj];
            Z += w;
            o += w * vsh[jj][lj];
        }
        if (t + tq < t1)
            attout[(size_t)(b * TT + t + tq) * INNERD + h * DHH + lj] = f2bf(o / Z);
        __syncthreads();                 // before qsh/ssh reuse
    }
}

// ---------------- host launcher ----------------
extern "C" void kernel_launch(void* const* d_in, const int* in_sizes, int n_in,
                              void* d_out, int out_size, void* d_ws, size_t ws_size,
                              hipStream_t stream) {
    const float* x      = (const float*)d_in[0];
    const float* media  = (const float*)d_in[1];
    const unsigned char* mloc = (const unsigned char*)d_in[2];   // jnp bool -> 1 byte
    const float* gamma  = (const float*)d_in[3];
    const float* beta   = (const float*)d_in[4];
    const float* Wq     = (const float*)d_in[5];
    const float* Wkv    = (const float*)d_in[6];
    const float* Wout   = (const float*)d_in[7];
    float* out          = (float*)d_out;

    char* ws = (char*)d_ws;
    bf16* xn     = (bf16*)ws;  ws += (size_t)BB * TT * DD * 2;          // 32 MB
    bf16* WqT    = (bf16*)ws;  ws += (size_t)INNERD * DD * 2;           // 4 MB (1024x2048)
    bf16* WkvT   = (bf16*)ws;  ws += (size_t)2 * INNERD * DVV * 2;      // 4 MB (2048x1024)
    bf16* WoutT  = (bf16*)ws;  ws += (size_t)DD * INNERD * 2;           // 4 MB (2048x1024)
    bf16* mediab = (bf16*)ws;  ws += (size_t)BB * JJ * DVV * 2;         // 4 MB
    float* qb    = (float*)ws; ws += (size_t)BB * TT * INNERD * 4;      // 32 MB
    float* kvb   = (float*)ws; ws += (size_t)BB * JJ * 2 * INNERD * 4;  // 16 MB
    bf16* attout = (bf16*)ws;  ws += (size_t)BB * TT * INNERD * 2;      // 16 MB
    int* starts  = (int*)ws;                                            // B*10 ints

    // 1) LayerNorm + bf16 cast
    ln_rows_kernel<<<BB * TT, 256, 0, stream>>>(x, gamma, beta, xn);

    // 2) weight transposes -> bf16 (N x K layout for WMMA B operand)
    transpose_f32_to_bf16_kernel<<<dim3(INNERD / 32, DD / 32), 256, 0, stream>>>(Wq, WqT, DD, INNERD);
    transpose_f32_to_bf16_kernel<<<dim3(2 * INNERD / 32, DVV / 32), 256, 0, stream>>>(Wkv, WkvT, DVV, 2 * INNERD);
    transpose_f32_to_bf16_kernel<<<dim3(DD / 32, INNERD / 32), 256, 0, stream>>>(Wout, WoutT, INNERD, DD);

    // 3) media -> bf16
    {
        int n = BB * JJ * DVV;
        cvt_bf16_kernel<<<n / 1024, 256, 0, stream>>>(media, mediab, n);
    }

    // 4) media-location block starts
    starts_kernel<<<BB, 32, 0, stream>>>(mloc, starts);

    // 5) q = xn @ Wq : (8192x2048)(2048x1024)
    gemm_bf16_wmma_kernel<<<dim3(INNERD / 128, (BB * TT) / 128), 256, 0, stream>>>(
        xn, WqT, qb, BB * TT, INNERD, DD);

    // 6) kv = media @ Wkv : (2048x1024)(1024x2048)
    gemm_bf16_wmma_kernel<<<dim3(2 * INNERD / 128, (BB * JJ) / 128), 256, 0, stream>>>(
        mediab, WkvT, kvb, BB * JJ, 2 * INNERD, DVV);

    // 7) masked cross-attention -> bf16 (zeros where no preceding media)
    flamingo_attn_kernel<<<dim3(HH, 9, BB), 256, 0, stream>>>(qb, kvb, starts, attout);

    // 8) out = attout @ Wout : (8192x1024)(1024x2048) -> d_out fp32
    gemm_bf16_wmma_kernel<<<dim3(DD / 128, (BB * TT) / 128), 256, 0, stream>>>(
        attout, WoutT, out, BB * TT, DD, INNERD);
}
